// CDEODE_51951924412751
// MI455X (gfx1250) — compile-verified, matrix operand order checked by reading, not run on hardware
//
#include <hip/hip_runtime.h>
#include <hip/hip_bf16.h>
#include <math.h>

typedef __attribute__((ext_vector_type(16))) _Float16 v16h;
typedef __attribute__((ext_vector_type(8)))  float    v8f;
typedef __attribute__((ext_vector_type(4)))  float    f4v;

#define BATCH  1024
#define LSTEPS 100
#define HID    128
#define CTRL   60
#define TRAIN  100
#define ROWS   16
#define NWG    (BATCH / ROWS)

// f16 weight workspace layout (element offsets)
#define S_FW2  0
#define S_FW3  16384
#define S_OW2  32768
#define S_OW3  49152
#define S_FW1  65536   // padded 128x32
#define S_OW1  69632   // padded 128x32
#define S_FW4  73728   // padded 16x128
#define S_OW4  75776   // padded 16x128

// Tsit5 tableau
__constant__ float d_TA[6][5] = {
  {0.f, 0.f, 0.f, 0.f, 0.f},
  {0.161f, 0.f, 0.f, 0.f, 0.f},
  {-0.008480655492356989f, 0.335480655492357f, 0.f, 0.f, 0.f},
  {2.8971530571054935f, -6.359448489975075f, 4.3622954328695815f, 0.f, 0.f},
  {5.325864828439257f, -11.748883564062828f, 7.4955393428898365f, -0.09249506636175525f, 0.f},
  {5.86145544294642f, -12.92096931784711f, 8.159367898576159f, -0.071584973281401f, -0.028269050394068383f}
};
__constant__ float d_TB[6] = {0.09646076681806523f, 0.01f, 0.4798896504144996f,
                              1.379008574103742f, -3.290069515436081f, 2.324710524099774f};
__constant__ float d_TC[6] = {0.f, 0.161f, 0.327f, 0.9f, 0.9800255409045097f, 1.f};

// Branch-free tanh: hardware v_tanh_f32 if available, else exp2+rcp identity.
#if __has_builtin(__builtin_amdgcn_tanhf)
__device__ __forceinline__ float tanh_fast(float x) { return __builtin_amdgcn_tanhf(x); }
#else
__device__ __forceinline__ float tanh_fast(float x) {
  float e = __builtin_amdgcn_exp2f(2.88539008177793f * x);   // e^(2x)
  return 1.f - 2.f * __builtin_amdgcn_rcpf(e + 1.f);
}
#endif

// Convert/pad all weight matrices to f16 B-operand layout source (row-major).
__global__ __launch_bounds__(256) void cde_prep(
    const float* __restrict__ fW1, const float* __restrict__ fW2,
    const float* __restrict__ fW3, const float* __restrict__ fW4,
    const float* __restrict__ oW1, const float* __restrict__ oW2,
    const float* __restrict__ oW3, const float* __restrict__ oW4,
    _Float16* __restrict__ wh) {
  int i = blockIdx.x * blockDim.x + threadIdx.x;
  if (i < HID * HID) {
    wh[S_FW2 + i] = (_Float16)fW2[i];
    wh[S_FW3 + i] = (_Float16)fW3[i];
    wh[S_OW2 + i] = (_Float16)oW2[i];
    wh[S_OW3 + i] = (_Float16)oW3[i];
  }
  if (i < HID * 32) {               // W1 padded [128][32]
    int j = i >> 5, k = i & 31;
    wh[S_FW1 + i] = (_Float16)((k < 3) ? fW1[j * 3 + k] : 0.f);
    wh[S_OW1 + i] = (_Float16)((k < 4) ? oW1[j * 4 + k] : 0.f);
  }
  if (i < 16 * HID) {               // W4 padded [16][128]
    int o = i >> 7, k = i & 127;
    wh[S_FW4 + i] = (_Float16)((o < 9) ? fW4[o * HID + k] : 0.f);
    wh[S_OW4 + i] = (_Float16)((o < 3) ? oW4[o * HID + k] : 0.f);
  }
}

__global__ __launch_bounds__(256) void cde_main(
    const float* __restrict__ ts, const float* __restrict__ ys,
    const float* __restrict__ fb1, const float* __restrict__ fb2,
    const float* __restrict__ fb3, const float* __restrict__ fb4,
    const float* __restrict__ ob1, const float* __restrict__ ob2,
    const float* __restrict__ ob3, const float* __restrict__ ob4,
    const _Float16* __restrict__ wh,
    float* __restrict__ out) {

  __shared__ __align__(16) _Float16 actIn[ROWS][32];     // padded MLP input tile
  __shared__ __align__(16) _Float16 act[2][ROWS][HID];   // hidden activation ping-pong
  __shared__ float sb1[HID], sb2[HID], sb3[HID];
  __shared__ float sb4[16];
  __shared__ float smlp[ROWS][16];                       // final-layer output tile (f32)
  __shared__ float sy[ROWS][3];                          // carried state
  __shared__ float sdx[ROWS][3];                         // control derivative (phase 1)
  __shared__ float sk[6][ROWS][3];                       // RK stage slopes

  const int t    = threadIdx.x;
  const int lane = t & 31;
  const int wv   = t >> 5;               // wave id == output N-tile
  const int m0   = blockIdx.x * ROWS;
  const int rr   = t / 3, dd = t % 3;    // RK-combine threads (t < 48)

  // B-operand fragment from global f16 weights, row-major [N][K] (B(k,n)=W[n][k]).
  auto load_bfrag = [&](const _Float16* base, int stride, int ntile, int kc) -> v16h {
    union { v16h v; f4v f[2]; } u;
    int n  = (ntile << 4) + (lane & 15);
    int hi = (lane >> 4) << 3;
    const _Float16* p = base + n * stride + (kc << 5) + hi;
    u.f[0] = *(const f4v*)(p);
    u.f[1] = *(const f4v*)(p + 16);
    return u.v;
  };
  // A fragment from LDS hidden activations (16 x 128 row-major).
  auto load_afrag = [&](int buf, int kc) -> v16h {
    union { v16h v; f4v f[2]; } u;
    const _Float16* p = &act[buf][lane & 15][(kc << 5) + ((lane >> 4) << 3)];
    u.f[0] = *(const f4v*)(p);
    u.f[1] = *(const f4v*)(p + 16);
    return u.v;
  };
  // A fragment from padded input tile (16 x 32).
  auto load_ain = [&]() -> v16h {
    union { v16h v; f4v f[2]; } u;
    const _Float16* p = &actIn[lane & 15][(lane >> 4) << 3];
    u.f[0] = *(const f4v*)(p);
    u.f[1] = *(const f4v*)(p + 16);
    return u.v;
  };
  // Hidden layer: 4 WMMA + bias + tanh -> LDS.
  auto wmma_layer = [&](int src, int dst, const v16h* wf, const float* bias) {
    v8f c = {};
#pragma unroll
    for (int kc = 0; kc < 4; ++kc) {
      v16h a = load_afrag(src, kc);
      c = __builtin_amdgcn_wmma_f32_16x16x32_f16(false, a, false, wf[kc],
                                                 (short)0, c, false, false);
    }
    int   n  = (wv << 4) + (lane & 15);
    float bn = bias[n];
    int   mb = (lane >> 4) << 3;
#pragma unroll
    for (int v = 0; v < 8; ++v)
      act[dst][mb + v][n] = (_Float16)tanh_fast(c[v] + bn);
  };

  // ---- phase-1 weights: biases to LDS, fragments to registers ----
  for (int i = t; i < HID; i += 256) { sb1[i] = fb1[i]; sb2[i] = fb2[i]; sb3[i] = fb3[i]; }
  if (t < 16) sb4[t] = (t < 9) ? fb4[t] : 0.f;
  for (int i = t; i < ROWS * 32; i += 256) actIn[i >> 5][i & 31] = (_Float16)0.f;  // zero pad once

  v16h w1f, w2f[4], w3f[4], w4f[4];
  w1f = load_bfrag(wh + S_FW1, 32, wv, 0);
#pragma unroll
  for (int kc = 0; kc < 4; ++kc) {
    w2f[kc] = load_bfrag(wh + S_FW2, HID, wv, kc);
    w3f[kc] = load_bfrag(wh + S_FW3, HID, wv, kc);
    w4f[kc] = load_bfrag(wh + S_FW4, HID, 0, kc);
  }

  // y0 = [ts[0], ys[:,0], ts[0]*ys[:,0]]; output col 0 = ys[:,0]
  if (t < ROWS) {
    int   b  = m0 + t;
    float yv = ys[b * LSTEPS + 0];
    float t0 = ts[0];
    sy[t][0] = t0; sy[t][1] = yv; sy[t][2] = t0 * yv;
    out[b * LSTEPS + 0] = yv;
  }
  __syncthreads();

  // ---- all-WMMA 4-layer MLP on the 16-row tile (input in actIn) ----
  auto mlp = [&]() {
    { // layer 1: padded 16x32 @ 32x128, one WMMA per wave
      v16h a = load_ain();
      v8f  c = {};
      c = __builtin_amdgcn_wmma_f32_16x16x32_f16(false, a, false, w1f,
                                                 (short)0, c, false, false);
      int   n  = (wv << 4) + (lane & 15);
      float bn = sb1[n];
      int   mb = (lane >> 4) << 3;
#pragma unroll
      for (int v = 0; v < 8; ++v)
        act[0][mb + v][n] = (_Float16)tanh_fast(c[v] + bn);
    }
    __syncthreads();
    wmma_layer(0, 1, w2f, sb2);
    __syncthreads();
    wmma_layer(1, 0, w3f, sb3);
    __syncthreads();
    if (wv == 0) { // layer 4: 16x128 @ 128x16 (padded), wave 0 only
      v8f c = {};
#pragma unroll
      for (int kc = 0; kc < 4; ++kc) {
        v16h a = load_afrag(0, kc);
        c = __builtin_amdgcn_wmma_f32_16x16x32_f16(false, a, false, w4f[kc],
                                                   (short)0, c, false, false);
      }
      int   n  = lane & 15;
      float bn = sb4[n];
      int   mb = (lane >> 4) << 3;
#pragma unroll
      for (int v = 0; v < 8; ++v)
        smlp[mb + v][n] = tanh_fast(c[v] + bn);
    }
    __syncthreads();
  };

  // ================= Phase 1: CDE steps (59) =================
  for (int s = 0; s < CTRL - 1; ++s) {
    float t0 = ts[s], t1 = ts[s + 1];
    float dt = t1 - t0;
    if (t < ROWS) {
      int   b   = m0 + t;
      float ya  = ys[b * LSTEPS + s], yb = ys[b * LSTEPS + s + 1];
      float inv = __builtin_amdgcn_rcpf(dt);
      sdx[t][0] = 1.f;
      sdx[t][1] = (yb - ya) * inv;
      sdx[t][2] = (t1 * yb - t0 * ya) * inv;
    }
    if (t < 48) actIn[rr][dd] = (_Float16)sy[rr][dd];
    __syncthreads();

    for (int st = 0; st < 6; ++st) {
      mlp();
      if (t < 48) {
        // k = mats(y) @ dx ; mats row-major 3x3 in smlp[r][0..8]
        float kk = smlp[rr][3 * dd + 0] * sdx[rr][0]
                 + smlp[rr][3 * dd + 1] * sdx[rr][1]
                 + smlp[rr][3 * dd + 2] * sdx[rr][2];
        sk[st][rr][dd] = kk;
        if (st < 5) {
          float acc = 0.f;
          for (int j = 0; j <= st; ++j) acc += d_TA[st + 1][j] * sk[j][rr][dd];
          actIn[rr][dd] = (_Float16)(sy[rr][dd] + dt * acc);
        }
      }
      __syncthreads();
    }
    if (t < 48) {
      float acc = 0.f;
      for (int j = 0; j < 6; ++j) acc += d_TB[j] * sk[j][rr][dd];
      float yn = sy[rr][dd] + dt * acc;
      sy[rr][dd] = yn;
      if (dd == 1) out[(m0 + rr) * LSTEPS + s + 1] = yn;
    }
    __syncthreads();
  }

  // ---- swap to phase-2 (ODE) weights ----
  for (int i = t; i < HID; i += 256) { sb1[i] = ob1[i]; sb2[i] = ob2[i]; sb3[i] = ob3[i]; }
  if (t < 16) sb4[t] = (t < 3) ? ob4[t] : 0.f;
  w1f = load_bfrag(wh + S_OW1, 32, wv, 0);
#pragma unroll
  for (int kc = 0; kc < 4; ++kc) {
    w2f[kc] = load_bfrag(wh + S_OW2, HID, wv, kc);
    w3f[kc] = load_bfrag(wh + S_OW3, HID, wv, kc);
    w4f[kc] = load_bfrag(wh + S_OW4, HID, 0, kc);
  }
  __syncthreads();

  // ================= Phase 2: ODE steps (40) =================
  for (int s = 0; s < TRAIN - CTRL; ++s) {
    float t0 = ts[CTRL - 1 + s];
    float dt = ts[CTRL + s] - t0;
    if (t < 48) {
      actIn[rr][dd] = (_Float16)sy[rr][dd];
      if (dd == 0) actIn[rr][3] = (_Float16)t0;       // stage-1 time (c1 = 0)
    }
    __syncthreads();

    for (int st = 0; st < 6; ++st) {
      mlp();
      if (t < 48) {
        sk[st][rr][dd] = smlp[rr][dd];
        if (st < 5) {
          float acc = 0.f;
          for (int j = 0; j <= st; ++j) acc += d_TA[st + 1][j] * sk[j][rr][dd];
          actIn[rr][dd] = (_Float16)(sy[rr][dd] + dt * acc);
          if (dd == 0) actIn[rr][3] = (_Float16)(t0 + d_TC[st + 1] * dt);
        }
      }
      __syncthreads();
    }
    if (t < 48) {
      float acc = 0.f;
      for (int j = 0; j < 6; ++j) acc += d_TB[j] * sk[j][rr][dd];
      float yn = sy[rr][dd] + dt * acc;
      sy[rr][dd] = yn;
      if (dd == 1) out[(m0 + rr) * LSTEPS + CTRL + s] = yn;
    }
    __syncthreads();
  }
}

extern "C" void kernel_launch(void* const* d_in, const int* in_sizes, int n_in,
                              void* d_out, int out_size, void* d_ws, size_t ws_size,
                              hipStream_t stream) {
  (void)in_sizes; (void)n_in; (void)out_size; (void)ws_size;
  const float* ts  = (const float*)d_in[0];
  const float* ys  = (const float*)d_in[1];
  const float* fW1 = (const float*)d_in[5];
  const float* fb1 = (const float*)d_in[6];
  const float* fW2 = (const float*)d_in[7];
  const float* fb2 = (const float*)d_in[8];
  const float* fW3 = (const float*)d_in[9];
  const float* fb3 = (const float*)d_in[10];
  const float* fW4 = (const float*)d_in[11];
  const float* fb4 = (const float*)d_in[12];
  const float* oW1 = (const float*)d_in[13];
  const float* ob1 = (const float*)d_in[14];
  const float* oW2 = (const float*)d_in[15];
  const float* ob2 = (const float*)d_in[16];
  const float* oW3 = (const float*)d_in[17];
  const float* ob3 = (const float*)d_in[18];
  const float* oW4 = (const float*)d_in[19];
  const float* ob4 = (const float*)d_in[20];

  _Float16* wh = (_Float16*)d_ws;   // ~152 KB of f16 weights

  cde_prep<<<(HID * HID + 255) / 256, 256, 0, stream>>>(fW1, fW2, fW3, fW4,
                                                        oW1, oW2, oW3, oW4, wh);
  cde_main<<<NWG, 256, 0, stream>>>(ts, ys, fb1, fb2, fb3, fb4,
                                    ob1, ob2, ob3, ob4, wh, (float*)d_out);
}
